// BaseCausalWanSelfAttention_22445499089790
// MI455X (gfx1250) — matrix operand order repaired and checked by reading, not compiled, not run on hardware
//
#include <hip/hip_runtime.h>

// Problem constants (match reference)
#define Bb   2
#define Ss   3072
#define Hh   12
#define Dd   128
#define LW   1560
#define SINKN 128

typedef __attribute__((ext_vector_type(16))) _Float16 v16h;
typedef __attribute__((ext_vector_type(8)))  float    v8f;

// fold (1/sqrt(D)) * log2(e) into Q so softmax can use exp2
#define QSCALE (0.08838834764831845f * 1.4426950408889634f)

__device__ __forceinline__ int imax(int a, int b) { return a > b ? a : b; }
__device__ __forceinline__ int imin(int a, int b) { return a < b ? a : b; }

// ---------------------------------------------------------------------------
// 16-lane butterfly reductions in pure VALU via DPP16 (no LDS pipe).
// Each 16-lane DPP row reduces independently == our half-wave row groups.
//   xor1 = quad_perm(1,0,3,2)=0xB1, xor2 = quad_perm(2,3,0,1)=0x4E,
//   row_half_mirror=0x141 (combine 8-lane halves), row_mirror=0x140 (combine rows)
// ---------------------------------------------------------------------------
template <int CTRL>
__device__ __forceinline__ float dpp_bfly(float x) {
  int v = __builtin_bit_cast(int, x);
  int p = __builtin_amdgcn_update_dpp(v, v, CTRL, 0xF, 0xF, true);
  return __builtin_bit_cast(float, p);
}
__device__ __forceinline__ float row_max16(float x) {
  x = fmaxf(x, dpp_bfly<0xB1>(x));
  x = fmaxf(x, dpp_bfly<0x4E>(x));
  x = fmaxf(x, dpp_bfly<0x141>(x));
  x = fmaxf(x, dpp_bfly<0x140>(x));
  return x;
}
__device__ __forceinline__ float row_sum16(float x) {
  x += dpp_bfly<0xB1>(x);
  x += dpp_bfly<0x4E>(x);
  x += dpp_bfly<0x141>(x);
  x += dpp_bfly<0x140>(x);
  return x;
}

// ---------------------------------------------------------------------------
// Pre-pass 1: RoPE(Q), RoPE(K) -> f16, relayout [b][s][h][d] -> [b][h][s][d].
// Q additionally scaled by QSCALE.
// ---------------------------------------------------------------------------
__global__ void rope_qk_kernel(const float* __restrict__ q, const float* __restrict__ k,
                               const float* __restrict__ cs, const float* __restrict__ sn,
                               _Float16* __restrict__ qr, _Float16* __restrict__ kr) {
  const int d = threadIdx.x;        // 0..127
  const int s = blockIdx.x;
  const int h = blockIdx.y;
  const int b = blockIdx.z;
  const size_t in = (((size_t)b * Ss + s) * Hh + h) * Dd + d;
  const int j = d >> 1;
  const float c  = cs[s * (Dd / 2) + j];
  const float sv = sn[s * (Dd / 2) + j];
  const float qd = q[in], kd = k[in];
  const float qp = q[in ^ 1], kp = k[in ^ 1];   // rotary partner (d^1)
  float oq, ok;
  if ((d & 1) == 0) { oq = qd * c - qp * sv; ok = kd * c - kp * sv; }
  else              { oq = qd * c + qp * sv; ok = kd * c + kp * sv; }
  const size_t oi = (((size_t)b * Hh + h) * Ss + s) * Dd + d;
  qr[oi] = (_Float16)(oq * QSCALE);
  kr[oi] = (_Float16)ok;
}

// ---------------------------------------------------------------------------
// Pre-pass 2: V -> f16 transposed: vt[b][h][d][s]
// ---------------------------------------------------------------------------
__global__ void vtr_kernel(const float* __restrict__ v, _Float16* __restrict__ vt) {
  const int t = threadIdx.x;                 // 0..127 -> s within tile
  const int s = blockIdx.x * 128 + t;
  const int h = blockIdx.y, b = blockIdx.z;
  const float* src = v + (((size_t)b * Ss + s) * Hh + h) * Dd;
  _Float16* dst = vt + ((size_t)b * Hh + h) * (size_t)Dd * Ss + s;
#pragma unroll 4
  for (int d = 0; d < Dd; ++d) dst[(size_t)d * Ss] = (_Float16)src[d];
}

// ---------------------------------------------------------------------------
// Flash attention: 4 waves/block, wave owns 16 query rows, block owns 64.
// ---------------------------------------------------------------------------
__launch_bounds__(128, 1)
__global__ void attn_kernel(const _Float16* __restrict__ qr, const _Float16* __restrict__ kr,
                            const _Float16* __restrict__ vt, float* __restrict__ out) {
  __shared__ __align__(16) _Float16 pbuf[4][16][40];   // per-wave P staging, padded rows

  const int lane = threadIdx.x & 31;
  const int wave = threadIdx.x >> 5;
  const int n16  = lane & 15;        // column within 16-wide tile
  const int hi   = (lane >> 4) & 1;  // which half-wave
  const int b = blockIdx.z, h = blockIdx.y;
  const int q0 = blockIdx.x * 64 + wave * 16;

  const _Float16* qbase = qr + ((size_t)b * Hh + h) * (size_t)Ss * Dd;
  const _Float16* kbase = kr + ((size_t)b * Hh + h) * (size_t)Ss * Dd;
  const _Float16* vbase = vt + ((size_t)b * Hh + h) * (size_t)Dd * Ss;

  // ---- Load Q as WMMA A operands (16x32 f16 per chunk, 4 chunks over D=128)
  v16h aq[4];
  {
    const _Float16* qrow = qbase + (size_t)(q0 + n16) * Dd;
#pragma unroll
    for (int c = 0; c < 4; ++c) {
      const int d0 = 32 * c + 8 * hi;
      union { v16h v; uint4 u[2]; } t;
      t.u[0] = *(const uint4*)(qrow + d0);
      t.u[1] = *(const uint4*)(qrow + d0 + 16);
      aq[c] = t.v;
    }
  }

  v8f acc[8];
#pragma unroll
  for (int cc = 0; cc < 8; ++cc)
#pragma unroll
    for (int e = 0; e < 8; ++e) acc[cc][e] = 0.0f;

  float mrow[8], lrow[8];
#pragma unroll
  for (int r = 0; r < 8; ++r) { mrow[r] = -1e30f; lrow[r] = 0.0f; }

  auto process = [&](int kb, bool masked) {
    // ---- S = Q K^T for 32 keys: two 16x16 score tiles, 4 K-steps of 32 each
    v8f s0, s1;
#pragma unroll
    for (int e = 0; e < 8; ++e) { s0[e] = 0.0f; s1[e] = 0.0f; }
#pragma unroll
    for (int c = 0; c < 4; ++c) {
      const _Float16* k0 = kbase + (size_t)(kb + n16) * Dd + 32 * c + 8 * hi;
      const _Float16* k1 = kbase + (size_t)(kb + 16 + n16) * Dd + 32 * c + 8 * hi;
      union { v16h v; uint4 u[2]; } t0, t1;
      t0.u[0] = *(const uint4*)k0; t0.u[1] = *(const uint4*)(k0 + 16);
      t1.u[0] = *(const uint4*)k1; t1.u[1] = *(const uint4*)(k1 + 16);
      s0 = __builtin_amdgcn_wmma_f32_16x16x32_f16(false, aq[c], false, t0.v,
                                                  (short)0, s0, false, false);
      s1 = __builtin_amdgcn_wmma_f32_16x16x32_f16(false, aq[c], false, t1.v,
                                                  (short)0, s1, false, false);
    }

    // ---- online softmax (row = r + 8*hi, col = n16 in C layout)
    float p0[8], p1[8], mloc[8];
#pragma unroll
    for (int r = 0; r < 8; ++r) {
      float a = s0[r], c2 = s1[r];
      if (masked) {
        const int i  = q0 + r + 8 * hi;
        const int j0 = kb + n16, j1 = j0 + 16;
        const bool ok0 = (j0 <= i) && ((i - j0 < LW) || (j0 < SINKN));
        const bool ok1 = (j1 <= i) && ((i - j1 < LW) || (j1 < SINKN));
        a  = ok0 ? a  : -1e30f;
        c2 = ok1 ? c2 : -1e30f;
      }
      p0[r] = a; p1[r] = c2;
      mloc[r] = fmaxf(a, c2);
    }
#pragma unroll
    for (int r = 0; r < 8; ++r) mloc[r] = row_max16(mloc[r]);   // DPP16 butterfly
#pragma unroll
    for (int r = 0; r < 8; ++r) {
      const float mnew = fmaxf(mrow[r], mloc[r]);
      const float corr = exp2f(mrow[r] - mnew);
      mrow[r] = mnew;
      const float e0 = exp2f(p0[r] - mnew);
      const float e1 = exp2f(p1[r] - mnew);
      lrow[r] = lrow[r] * corr + e0 + e1;   // per-lane partial row sum
#pragma unroll
      for (int cc = 0; cc < 8; ++cc) acc[cc][r] *= corr;
      const int row = r + 8 * hi;
      pbuf[wave][row][n16]      = (_Float16)e0;
      pbuf[wave][row][n16 + 16] = (_Float16)e1;
    }

    // intra-wave: LDS is in-order per wave; clobber stops compiler reordering
    asm volatile("s_wait_dscnt 0" ::: "memory");

    // ---- P as A operand (16x32 f16)
    v16h ap;
    {
      const _Float16* prow = &pbuf[wave][n16][0];
      union { v16h v; uint4 u[2]; } t;
      t.u[0] = *(const uint4*)(prow + 8 * hi);
      t.u[1] = *(const uint4*)(prow + 8 * hi + 16);
      ap = t.v;
    }

    // ---- O += P V for 8 output d-chunks of 16
#pragma unroll
    for (int cc = 0; cc < 8; ++cc) {
      const _Float16* vp = vbase + (size_t)(16 * cc + n16) * Ss + kb + 8 * hi;
      union { v16h v; uint4 u[2]; } t;
      t.u[0] = *(const uint4*)vp; t.u[1] = *(const uint4*)(vp + 16);
      acc[cc] = __builtin_amdgcn_wmma_f32_16x16x32_f16(false, ap, false, t.v,
                                                       (short)0, acc[cc], false, false);
    }
  };

  const int qmin = blockIdx.x * 64;
  const int qmax = qmin + 63;
  const int kstart = imax(0, qmin - (LW - 1)) & ~31;

  // A 32-key tile needs NO mask iff  kb+31 <= qmin  (causal for every block row)
  // and  qmax-kb < LW  (window for every block row). Sink tiles below the window
  // start are always fully valid (j<128 and j<i guaranteed there).
  const int sink_end = imin(SINKN, kstart);
  const int kmid_lo  = imax(kstart, (qmax - (LW - 1) + 31) & ~31); // first fully-in-window tile
  const int kmid_hi  = qmin;                                       // first tile touching diagonal

  for (int kb = 0; kb < sink_end; kb += 32)      process(kb, false);  // sink, unmasked
  for (int kb = kstart; kb < kmid_lo; kb += 32)  process(kb, true);   // window edge
  for (int kb = kmid_lo; kb < kmid_hi; kb += 32) process(kb, false);  // bulk, unmasked
  for (int kb = kmid_hi; kb <= qmax; kb += 32)   process(kb, true);   // diagonal

  // ---- final row-sum reduction (DPP16) + normalize + store (f32 [b][s][h][d])
#pragma unroll
  for (int r = 0; r < 8; ++r) lrow[r] = row_sum16(lrow[r]);
#pragma unroll
  for (int r = 0; r < 8; ++r) {
    const float inv = 1.0f / lrow[r];
    const int i = q0 + r + 8 * hi;
    float* op = out + (((size_t)b * Ss + i) * Hh + h) * Dd;
#pragma unroll
    for (int cc = 0; cc < 8; ++cc) op[16 * cc + n16] = acc[cc][r] * inv;
  }
}

// ---------------------------------------------------------------------------
extern "C" void kernel_launch(void* const* d_in, const int* in_sizes, int n_in,
                              void* d_out, int out_size, void* d_ws, size_t ws_size,
                              hipStream_t stream) {
  const float* q  = (const float*)d_in[0];
  const float* k  = (const float*)d_in[1];
  const float* v  = (const float*)d_in[2];
  const float* cs = (const float*)d_in[3];
  const float* sn = (const float*)d_in[4];
  float* out = (float*)d_out;

  const size_t N = (size_t)Bb * Hh * Ss * Dd;   // 9.44M elems
  _Float16* qr = (_Float16*)d_ws;               // N f16
  _Float16* kr = qr + N;                        // N f16
  _Float16* vt = kr + N;                        // N f16  (total ~56.6 MB)

  dim3 g1(Ss, Hh, Bb);
  rope_qk_kernel<<<g1, 128, 0, stream>>>(q, k, cs, sn, qr, kr);

  dim3 g2(Ss / 128, Hh, Bb);
  vtr_kernel<<<g2, 128, 0, stream>>>(v, vt);

  dim3 g3(Ss / 64, Hh, Bb);
  attn_kernel<<<g3, 128, 0, stream>>>(qr, kr, vt, out);
}
